// SE3EquivariantStack_60730837566349
// MI455X (gfx1250) — matrix operand order repaired and checked by reading, not compiled, not run on hardware
//
#include <hip/hip_runtime.h>
#include <hip/hip_bf16.h>

#define NN 50000      // nodes
#define NE 400000     // edges (divisible by 16)
#define CD 128        // node dim
#define DPD 64        // edge dim
#define HD 256        // hidden
#define NL 4          // depth
#define MIN_DIM 320   // 2*CD + DPD
#define KT1 10        // MIN_DIM/32 k-tiles for GEMM1
#define NT1 16        // HD/16 n-tiles for GEMM1
#define KT2 8         // HD/32 k-tiles for GEMM2
#define NT2 8         // CD/16 n-tiles for GEMM2

typedef __attribute__((ext_vector_type(16))) __bf16 bf16x16;
typedef __attribute__((ext_vector_type(8)))  float  f32x8;

__device__ __forceinline__ float silu_f(float x) { return x / (1.f + __expf(-x)); }

__device__ __forceinline__ f32x8 wmma_bf16(bf16x16 a, bf16x16 b, f32x8 c) {
  return __builtin_amdgcn_wmma_f32_16x16x32_bf16(false, a, false, b, (short)0, c, false, false);
}

// Scatter one (row, k) element into A-fragment-swizzled LDS:
// layout [kt][lane][elem], lane = row + 16*half, km = k&31,
// half = (km>>3)&1, elem = ((km>>4)<<3)|(km&7)   (ISA 7.12.2 16-bit A 16x32)
__device__ __forceinline__ void store_swz(__bf16* base, int row, int k, float v) {
  const int kt   = k >> 5;
  const int km   = k & 31;
  const int slot = row + (((km >> 3) & 1) << 4);
  const int elem = ((km >> 4) << 3) | (km & 7);
  base[(kt * 32 + slot) * 16 + elem] = (__bf16)v;
}

// ---- weight fp32 -> bf16 fragment-major packing (once per launch) ------------
// Packed layout per layer: [(kt*NT + nt)*32 + lane][16 elems], where a lane's
// 16 elems are B[k = kt*32 + (elem>>3)*16 + (lane>>4)*8 + (elem&7)][n = nt*16 + (lane&15)].
__global__ __launch_bounds__(256)
void pack_kernel(const float* __restrict__ Wn1, const float* __restrict__ Wc1,
                 const float* __restrict__ Wn2,
                 unsigned short* __restrict__ Wn1p, unsigned short* __restrict__ Wc1p,
                 unsigned short* __restrict__ Wn2p) {
  const int i  = blockIdx.x * 256 + threadIdx.x;
  const int n1 = NL * KT1 * NT1 * 512;   // 327680
  const int n2 = NL * KT2 * NT2 * 512;   // 131072
  if (i < n1) {
    const int l = i / (KT1 * NT1 * 512);
    const int rem = i % (KT1 * NT1 * 512);
    const int elem = rem & 15;
    const int lane = (rem >> 4) & 31;
    const int nt   = (rem >> 9) & 15;
    const int kt   = rem >> 13;
    const int k = kt * 32 + ((elem >> 3) << 4) + ((lane >> 4) << 3) + (elem & 7);
    const int n = nt * 16 + (lane & 15);
    const size_t s = ((size_t)l * MIN_DIM + k) * HD + n;
    ((__bf16*)Wn1p)[i] = (__bf16)Wn1[s];
    ((__bf16*)Wc1p)[i] = (__bf16)Wc1[s];
  }
  if (i < n2) {
    const int l = i / (KT2 * NT2 * 512);
    const int rem = i % (KT2 * NT2 * 512);
    const int elem = rem & 15;
    const int lane = (rem >> 4) & 31;
    const int nt   = (rem >> 9) & 7;
    const int kt   = rem >> 12;
    const int k = kt * 32 + ((elem >> 3) << 4) + ((lane >> 4) << 3) + (elem & 7);
    const int n = nt * 16 + (lane & 15);
    ((__bf16*)Wn2p)[i] = (__bf16)Wn2[((size_t)l * HD + k) * CD + n];
  }
}

// ---- one EGNN layer: one wave32 per 16-edge tile -----------------------------
__global__ __launch_bounds__(32)
void egnn_edge_kernel(const float* __restrict__ h_in, const float* __restrict__ x_in,
                      float* __restrict__ h_out, float* __restrict__ x_out,
                      const int* __restrict__ src, const int* __restrict__ dst,
                      const float* __restrict__ edist,
                      const float* __restrict__ We1, const float* __restrict__ be1,
                      const float* __restrict__ We2, const float* __restrict__ be2,
                      const unsigned short* __restrict__ Wn1p_, const float* __restrict__ bn1,
                      const unsigned short* __restrict__ Wn2p_, const float* __restrict__ bn2,
                      const unsigned short* __restrict__ Wc1p_, const float* __restrict__ bc1,
                      const float* __restrict__ Wc2)
{
  const bf16x16* BpN1 = (const bf16x16*)Wn1p_;   // [KT1*NT1*32] fragments
  const bf16x16* BpC1 = (const bf16x16*)Wc1p_;
  const bf16x16* BpN2 = (const bf16x16*)Wn2p_;   // [KT2*NT2*32] fragments

  __shared__ bf16x16 miA[KT1][32];   // swizzled A tile of mi      (10 KB)
  __shared__ bf16x16 hidA[KT2][32];  // swizzled A tile of hidN    (8 KB)
  __shared__ __bf16  hidC[16][HD];   // row-major coord hidden     (8 KB)
  __bf16* miS  = (__bf16*)&miA[0][0];
  __bf16* hidS = (__bf16*)&hidA[0][0];

  const int lane = threadIdx.x;       // 0..31
  const int e0   = blockIdx.x * 16;   // NE % 16 == 0
  const int col  = lane & 15;
  const int half = lane >> 4;

  __builtin_prefetch(BpN1 + lane, 0, 1);
  __builtin_prefetch(BpC1 + lane, 0, 1);
  __builtin_prefetch(BpN2 + lane, 0, 1);

  // ---- gather h[src], h[dst] (float4 loads), scatter swizzled bf16 into LDS ----
  {
    const int r  = lane >> 1;
    const int c0 = (lane & 1) * 64;
    const int es = src[e0 + r];
    const int ed = dst[e0 + r];
    const float4* ps = (const float4*)(h_in + (size_t)es * CD + c0);
    const float4* pd = (const float4*)(h_in + (size_t)ed * CD + c0);
#pragma unroll 4
    for (int q = 0; q < 16; ++q) {
      const float4 a = ps[q];
      const float4 b = pd[q];
      const int k = c0 + q * 4;
      store_swz(miS, r, k + 0, a.x); store_swz(miS, r, k + 1, a.y);
      store_swz(miS, r, k + 2, a.z); store_swz(miS, r, k + 3, a.w);
      store_swz(miS, r, CD + k + 0, b.x); store_swz(miS, r, CD + k + 1, b.y);
      store_swz(miS, r, CD + k + 2, b.z); store_swz(miS, r, CD + k + 3, b.w);
    }
  }

  // ---- edge MLP (scalar -> 64 -> 64), lanes 0..15, one edge each ----
  if (lane < 16) {
    const float d = edist[e0 + lane];
    float h1[DPD];
#pragma unroll
    for (int j = 0; j < DPD; ++j) h1[j] = silu_f(d * We1[j] + be1[j]);
    for (int k = 0; k < DPD; ++k) {
      float acc = be2[k];
#pragma unroll 8
      for (int j = 0; j < DPD; ++j) acc += h1[j] * We2[j * DPD + k];
      store_swz(miS, lane, 2 * CD + k, acc);
    }
  }
  __syncthreads();

  // ---- hoist all 10 A fragments for GEMM1 (2x ds_load_b128 each) ----
  bf16x16 afr[KT1];
#pragma unroll
  for (int kt = 0; kt < KT1; ++kt) afr[kt] = miA[kt][lane];

  // ---- GEMM1: mi[16,320] @ {Wn1,Wc1}[320,256] (+bias, fused SiLU) ----
  for (int nt = 0; nt < NT1; ++nt) {
    const float bN = bn1[nt * 16 + col];
    const float bC = bc1[nt * 16 + col];
    f32x8 accN = {bN, bN, bN, bN, bN, bN, bN, bN};
    f32x8 accC = {bC, bC, bC, bC, bC, bC, bC, bC};
#pragma unroll
    for (int kt = 0; kt < KT1; ++kt) {
      accN = wmma_bf16(afr[kt], BpN1[(kt * NT1 + nt) * 32 + lane], accN);
      accC = wmma_bf16(afr[kt], BpC1[(kt * NT1 + nt) * 32 + lane], accC);
    }
#pragma unroll
    for (int j = 0; j < 8; ++j) {
      const int m = j + (half << 3);
      const int n = nt * 16 + col;
      store_swz(hidS, m, n, silu_f(accN[j]));   // swizzled for GEMM2 A
      hidC[m][n] = (__bf16)silu_f(accC[j]);
    }
  }
  __syncthreads();

  // ---- hoist 8 A fragments for GEMM2 ----
  bf16x16 hfr[KT2];
#pragma unroll
  for (int kt = 0; kt < KT2; ++kt) hfr[kt] = hidA[kt][lane];
  int dnj[8];
#pragma unroll
  for (int j = 0; j < 8; ++j) dnj[j] = dst[e0 + j + (half << 3)];

  // ---- GEMM2: hidN[16,256] @ Wn2[256,128] + bn2 ; scatter-add into h_out[dst] ----
  for (int nt = 0; nt < NT2; ++nt) {
    const float b0 = bn2[nt * 16 + col];
    f32x8 acc = {b0, b0, b0, b0, b0, b0, b0, b0};
#pragma unroll
    for (int kt = 0; kt < KT2; ++kt)
      acc = wmma_bf16(hfr[kt], BpN2[(kt * NT2 + nt) * 32 + lane], acc);
#pragma unroll
    for (int j = 0; j < 8; ++j)
      atomicAdd(&h_out[(size_t)dnj[j] * CD + nt * 16 + col], acc[j]);
  }

  // ---- coord weight cw = hidC . Wc2 ; x_out[dst] += cw*(x[src]-x[dst]) ----
  if (lane < 16) {
    float cw = 0.f;
#pragma unroll 8
    for (int k = 0; k < HD; ++k) cw += (float)hidC[lane][k] * Wc2[k];
    const int e  = e0 + lane;
    const int es = src[e];
    const int ed = dst[e];
#pragma unroll
    for (int d = 0; d < 3; ++d) {
      const float dx = x_in[es * 3 + d] - x_in[ed * 3 + d];
      atomicAdd(&x_out[ed * 3 + d], cw * dx);
    }
  }
}

// ---- final LayerNorm over C=128, one wave per node ---------------------------
__global__ __launch_bounds__(256)
void ln_kernel(const float* __restrict__ h, const float* __restrict__ gamma,
               const float* __restrict__ beta, float* __restrict__ out) {
  const int lane = threadIdx.x & 31;
  const int node = blockIdx.x * 8 + (threadIdx.x >> 5);
  if (node >= NN) return;
  const float* row = h + (size_t)node * CD;
  float v0 = row[lane], v1 = row[lane + 32], v2 = row[lane + 64], v3 = row[lane + 96];
  float s = v0 + v1 + v2 + v3;
#pragma unroll
  for (int o = 16; o > 0; o >>= 1) s += __shfl_xor(s, o, 32);
  const float mu = s * (1.f / 128.f);
  const float t0 = v0 - mu, t1 = v1 - mu, t2 = v2 - mu, t3 = v3 - mu;
  float vs = t0 * t0 + t1 * t1 + t2 * t2 + t3 * t3;
#pragma unroll
  for (int o = 16; o > 0; o >>= 1) vs += __shfl_xor(vs, o, 32);
  const float inv = rsqrtf(vs * (1.f / 128.f) + 1e-5f);
  out[(size_t)node * CD + lane]      = gamma[lane]      * t0 * inv + beta[lane];
  out[(size_t)node * CD + lane + 32] = gamma[lane + 32] * t1 * inv + beta[lane + 32];
  out[(size_t)node * CD + lane + 64] = gamma[lane + 64] * t2 * inv + beta[lane + 64];
  out[(size_t)node * CD + lane + 96] = gamma[lane + 96] * t3 * inv + beta[lane + 96];
}

extern "C" void kernel_launch(void* const* d_in, const int* in_sizes, int n_in,
                              void* d_out, int out_size, void* d_ws, size_t ws_size,
                              hipStream_t stream) {
  (void)in_sizes; (void)n_in; (void)out_size; (void)ws_size;
  const float* single = (const float*)d_in[0];
  /* d_in[1] 'pair' unused */
  const float* coords = (const float*)d_in[2];
  const int*   eidx   = (const int*)d_in[3];     // [2, E]: src row, then dst row
  const float* edist  = (const float*)d_in[4];
  const float* We1 = (const float*)d_in[5];
  const float* be1 = (const float*)d_in[6];
  const float* We2 = (const float*)d_in[7];
  const float* be2 = (const float*)d_in[8];
  const float* Wn1 = (const float*)d_in[9];
  const float* bn1 = (const float*)d_in[10];
  const float* Wn2 = (const float*)d_in[11];
  const float* bn2 = (const float*)d_in[12];
  const float* Wc1 = (const float*)d_in[13];
  const float* bc1 = (const float*)d_in[14];
  const float* Wc2 = (const float*)d_in[15];
  const float* gamma = (const float*)d_in[16];
  const float* beta  = (const float*)d_in[17];

  char* ws = (char*)d_ws;
  size_t off = 0;
  auto carve = [&](size_t bytes) -> void* {
    void* p = (void*)(ws + off);
    off += (bytes + (size_t)255) & ~(size_t)255;
    return p;
  };
  float* h0 = (float*)carve(sizeof(float) * (size_t)NN * CD);
  float* h1 = (float*)carve(sizeof(float) * (size_t)NN * CD);
  float* x0 = (float*)carve(sizeof(float) * (size_t)NN * 3);
  float* x1 = (float*)carve(sizeof(float) * (size_t)NN * 3);
  unsigned short* Wn1p = (unsigned short*)carve(2ull * NL * MIN_DIM * HD);
  unsigned short* Wc1p = (unsigned short*)carve(2ull * NL * MIN_DIM * HD);
  unsigned short* Wn2p = (unsigned short*)carve(2ull * NL * HD * CD);

  // 1) pack weights into bf16 fragment-major layout (once per launch)
  pack_kernel<<<(NL * MIN_DIM * HD + 255) / 256, 256, 0, stream>>>(Wn1, Wc1, Wn2, Wn1p, Wc1p, Wn2p);
  // 2) init double buffers
  hipMemcpyAsync(h0, single, sizeof(float) * (size_t)NN * CD, hipMemcpyDeviceToDevice, stream);
  hipMemcpyAsync(x0, coords, sizeof(float) * (size_t)NN * 3, hipMemcpyDeviceToDevice, stream);

  float* hc = h0; float* hn = h1;
  float* xc = x0; float* xn = x1;
  for (int l = 0; l < NL; ++l) {
    hipMemcpyAsync(hn, hc, sizeof(float) * (size_t)NN * CD, hipMemcpyDeviceToDevice, stream);
    hipMemcpyAsync(xn, xc, sizeof(float) * (size_t)NN * 3, hipMemcpyDeviceToDevice, stream);
    egnn_edge_kernel<<<NE / 16, 32, 0, stream>>>(
        hc, xc, hn, xn,
        eidx, eidx + NE, edist,
        We1 + (size_t)l * DPD, be1 + (size_t)l * DPD,
        We2 + (size_t)l * DPD * DPD, be2 + (size_t)l * DPD,
        Wn1p + (size_t)l * MIN_DIM * HD, bn1 + (size_t)l * HD,
        Wn2p + (size_t)l * HD * CD,      bn2 + (size_t)l * CD,
        Wc1p + (size_t)l * MIN_DIM * HD, bc1 + (size_t)l * HD,
        Wc2 + (size_t)l * HD);
    float* t;
    t = hc; hc = hn; hn = t;
    t = xc; xc = xn; xn = t;
  }
  // 3) final LayerNorm -> d_out[0 : N*C), then coords -> d_out[N*C : N*C + N*3)
  ln_kernel<<<NN / 8, 256, 0, stream>>>(hc, gamma, beta, (float*)d_out);
  hipMemcpyAsync((float*)d_out + (size_t)NN * CD, xc, sizeof(float) * (size_t)NN * 3,
                 hipMemcpyDeviceToDevice, stream);
}
